// Head_31250182046596
// MI455X (gfx1250) — matrix operand order repaired and checked by reading, not compiled, not run on hardware
//
#include <hip/hip_runtime.h>

// Causal single-head attention, MI455X (gfx1250), wave32 + WMMA f16.
// B=4, T=4096, C=1024, HS=128.  fp32 in/out, f16 WMMA operands, f32 accum.
//
// Round-3 change: __builtin_amdgcn_sched_barrier(0) fences between the
// fragment-load batches and the WMMA chains. Round 2 showed the post-RA
// scheduler sinking loads back to their consumers (ds_load -> s_wait_dscnt 0
// -> wmma, fully serialized); the barrier pins the batched issue order so the
// 8 fragments stay live in distinct VGPRs and waits become partial.

typedef __attribute__((ext_vector_type(16))) _Float16 v16h;
typedef __attribute__((ext_vector_type(8)))  _Float16 v8h;
typedef __attribute__((ext_vector_type(8)))  float    v8f;
typedef __attribute__((ext_vector_type(4)))  float    v4f;

#define B_  4
#define T_  4096
#define C_  1024
#define HS_ 128
#define M_  (B_ * T_)   // 16384 total rows

static __device__ __forceinline__ v8f wmma_f16(v16h a, v16h b, v8f c) {
  // 8 args: (neg_a, A, neg_b, B, c_mod, C, reuse_a, reuse_b)
  return __builtin_amdgcn_wmma_f32_16x16x32_f16(false, a, false, b, (short)0, c,
                                                false, false);
}

// ---------------------------------------------------------------------------
// Projection GEMM:  out = (x[M,C] @ W[C,HS]) * scale, stored as f16.
//   transposed==0 : out row-major  [M, HS]
//   transposed==1 : out transposed [HS, M]   (for V, so attn B-frags are contig)
// Block = 256 threads = 8 waves. Each wave: 16 rows x 128 cols (8 WMMA accums).
// W k-slab (32 x 128) staged transposed in LDS as f16.
// ---------------------------------------------------------------------------
__global__ void __launch_bounds__(256)
proj_gemm_kernel(const float* __restrict__ x, const float* __restrict__ W,
                 _Float16* __restrict__ out, float scale, int transposed)
{
  __shared__ _Float16 Wt[128][32];     // Wt[n][k] for the current k-slab

  const int tid  = threadIdx.x;
  const int wave = tid >> 5;
  const int lane = tid & 31;
  const int m    = lane & 15;          // A-row / B-col within tile
  const int hh   = lane >> 4;          // lane-half
  const int rowBase = blockIdx.x * 128 + wave * 16;
  const float* xrow = x + (size_t)(rowBase + m) * C_;

  v8f acc[8] = {};

  for (int kk = 0; kk < C_; kk += 32) {
    __syncthreads();
    {   // cooperative W[kk..kk+32) x [0..128) -> LDS (transposed, f32->f16)
      const int r  = tid >> 3;         // 0..31 : k-row within slab
      const int c0 = (tid & 7) * 16;   // 16-col group
      const float* wp = W + (size_t)(kk + r) * HS_ + c0;
#pragma unroll
      for (int i = 0; i < 16; i += 4) {
        v4f w4 = *(const v4f*)(wp + i);
#pragma unroll
        for (int q = 0; q < 4; ++q) Wt[c0 + i + q][r] = (_Float16)w4[q];
      }
    }
    __syncthreads();

    // A fragment (16x32 f16): lane m, half hh; e -> k = hh*16 + e
    v16h a;
    {
      const float* p0 = xrow + kk + hh * 16;       // 32B aligned
      const float* p1 = xrow + kk + hh * 16 + 8;
      v4f a0 = *(const v4f*)(p0);
      v4f a1 = *(const v4f*)(p0 + 4);
      v4f a2 = *(const v4f*)(p1);
      v4f a3 = *(const v4f*)(p1 + 4);
#pragma unroll
      for (int q = 0; q < 4; ++q) {
        a[q]      = (_Float16)a0[q];
        a[4 + q]  = (_Float16)a1[q];
        a[8 + q]  = (_Float16)a2[q];
        a[12 + q] = (_Float16)a3[q];
      }
    }

    // Batch ALL B-fragment LDS loads, fence, then WMMA chain back-to-back.
    v16h bfr[8];
#pragma unroll
    for (int nt = 0; nt < 8; ++nt) {
      // B fragment (32x16): lane n = nt*16+m, half hh; e -> k = hh*16 + e
      const _Float16* wb = &Wt[nt * 16 + m][hh * 16];
      v8h b0 = *(const v8h*)(wb);
      v8h b1 = *(const v8h*)(wb + 8);
#pragma unroll
      for (int q = 0; q < 8; ++q) { bfr[nt][q] = b0[q]; bfr[nt][8 + q] = b1[q]; }
    }
    __builtin_amdgcn_sched_barrier(0);   // keep loads issued before WMMAs
#pragma unroll
    for (int nt = 0; nt < 8; ++nt)
      acc[nt] = wmma_f16(a, bfr[nt], acc[nt]);
  }

  // C/D layout: lane col n = l%16, VGPR r -> row m = r + 8*(l/16)
  if (!transposed) {
#pragma unroll
    for (int nt = 0; nt < 8; ++nt)
#pragma unroll
      for (int r = 0; r < 8; ++r)
        out[(size_t)(rowBase + 8 * hh + r) * HS_ + nt * 16 + m] =
            (_Float16)(acc[nt][r] * scale);
  } else {
#pragma unroll
    for (int nt = 0; nt < 8; ++nt) {
      v8h o;
#pragma unroll
      for (int r = 0; r < 8; ++r) o[r] = (_Float16)(acc[nt][r] * scale);
      // rows r are contiguous in the transposed layout -> one 16B store
      *(v8h*)(out + (size_t)(nt * 16 + m) * M_ + rowBase + 8 * hh) = o;
    }
  }
}

// ---------------------------------------------------------------------------
// Flash attention: each wave owns 16 query rows; loops over 32-key blocks.
// Qh, Kh row-major f16 [M, HS] (Q pre-scaled by HS^-0.5); Vt transposed
// f16 [HS, M]. K+V f16 = 8 MB total -> resident in the 192 MB L2, so direct
// per-wave global loads already stream at L2 bandwidth (no LDS staging).
// ---------------------------------------------------------------------------
__global__ void __launch_bounds__(256)
attn_kernel(const _Float16* __restrict__ Qh, const _Float16* __restrict__ Kh,
            const _Float16* __restrict__ Vt, float* __restrict__ out)
{
  __shared__ _Float16 Pbuf[8][16 * 32];   // per-wave P bounce buffer (8KB)

  const int tid  = threadIdx.x;
  const int wave = tid >> 5;
  const int lane = tid & 31;
  const int m    = lane & 15;
  const int hh   = lane >> 4;

  const int b  = blockIdx.x >> 5;                 // 4 batches * 32 q-blocks
  const int q0 = (blockIdx.x & 31) * 128 + wave * 16;
  const size_t rowQ = (size_t)b * T_ + q0;

  // Resident Q fragments: 4 x (16x32) covering HS=128
  v16h qf[4];
  {
    const _Float16* qrow = Qh + (rowQ + m) * HS_;
#pragma unroll
    for (int kb = 0; kb < 4; ++kb) {
      v8h qa = *(const v8h*)(qrow + kb * 32 + hh * 16);
      v8h qb = *(const v8h*)(qrow + kb * 32 + hh * 16 + 8);
#pragma unroll
      for (int i = 0; i < 8; ++i) { qf[kb][i] = qa[i]; qf[kb][8 + i] = qb[i]; }
    }
  }

  float mstate[8], lstate[8];
  v8f acc[8] = {};
#pragma unroll
  for (int r = 0; r < 8; ++r) { mstate[r] = -3.0e38f; lstate[r] = 0.0f; }

  _Float16* P = &Pbuf[wave][0];
  const int  kmax = q0 + 16;             // causal bound for this wave's rows

  for (int k0 = 0; k0 < kmax; k0 += 32) {
    const _Float16* krow0 = Kh + ((size_t)b * T_ + k0 + m) * HS_;
    const _Float16* krow1 = krow0 + (size_t)16 * HS_;
    __builtin_prefetch(krow0 + (size_t)32 * HS_, 0, 0);   // next key block

    // ---- Batch all 8 K-fragment loads (B 32x16: lane n=m, k = hh*16+e)
    v16h kf0[4], kf1[4];
#pragma unroll
    for (int kb = 0; kb < 4; ++kb) {
      kf0[kb] = *(const v16h*)(krow0 + kb * 32 + hh * 16);  // 32B aligned
      kf1[kb] = *(const v16h*)(krow1 + kb * 32 + hh * 16);
    }
    __builtin_amdgcn_sched_barrier(0);   // loads stay batched ahead of WMMAs

    // ---- S = Q Kᵀ : two 16x16 f32 tiles, WMMAs back-to-back
    v8f s0 = {}, s1 = {};
#pragma unroll
    for (int kb = 0; kb < 4; ++kb) {
      s0 = wmma_f16(qf[kb], kf0[kb], s0);
      s1 = wmma_f16(qf[kb], kf1[kb], s1);
    }

    // ---- Issue V-fragment loads NOW: softmax VALU/exp work below hides
    //      their latency (addresses are independent of softmax results).
    v16h vf[8];
#pragma unroll
    for (int j = 0; j < 8; ++j) {
      const _Float16* vptr = Vt + (size_t)(j * 16 + m) * M_ +
                             (size_t)b * T_ + k0 + hh * 16;
      vf[j] = *(const v16h*)vptr;                  // 32B aligned
    }
    __builtin_amdgcn_sched_barrier(0);   // V loads issue before softmax work

    // ---- causal mask + row max (rows span 16 lanes of this half-wave)
    const int qi0 = q0 + 8 * hh;
    float rmax[8];
#pragma unroll
    for (int r = 0; r < 8; ++r) {
      const int qi = qi0 + r;
      if (k0 + m > qi)      s0[r] = -3.0e38f;
      if (k0 + 16 + m > qi) s1[r] = -3.0e38f;
      rmax[r] = fmaxf(s0[r], s1[r]);
    }
#pragma unroll
    for (int off = 1; off <= 8; off <<= 1)
#pragma unroll
      for (int r = 0; r < 8; ++r)
        rmax[r] = fmaxf(rmax[r], __shfl_xor(rmax[r], off, 32));

    // ---- online softmax update
    float alpha[8], rsum[8];
#pragma unroll
    for (int r = 0; r < 8; ++r) {
      const float mn = fmaxf(mstate[r], rmax[r]);
      alpha[r]  = __expf(mstate[r] - mn);
      mstate[r] = mn;
      s0[r] = __expf(s0[r] - mn);
      s1[r] = __expf(s1[r] - mn);
      rsum[r] = s0[r] + s1[r];
    }
#pragma unroll
    for (int off = 1; off <= 8; off <<= 1)
#pragma unroll
      for (int r = 0; r < 8; ++r)
        rsum[r] += __shfl_xor(rsum[r], off, 32);
#pragma unroll
    for (int r = 0; r < 8; ++r)
      lstate[r] = lstate[r] * alpha[r] + rsum[r];

    // ---- P: C-layout -> A-layout via per-wave LDS (same-wave DS is in-order)
#pragma unroll
    for (int r = 0; r < 8; ++r) {
      P[(8 * hh + r) * 32 + m]      = (_Float16)s0[r];
      P[(8 * hh + r) * 32 + 16 + m] = (_Float16)s1[r];
    }
#pragma unroll
    for (int j = 0; j < 8; ++j)
#pragma unroll
      for (int r = 0; r < 8; ++r)
        acc[j][r] *= alpha[r];

    v16h pa;
    {
      const _Float16* prow = P + m * 32;
      v8h p0v = *(const v8h*)(prow + hh * 16);       // 16B aligned
      v8h p1v = *(const v8h*)(prow + hh * 16 + 8);
#pragma unroll
      for (int i = 0; i < 8; ++i) { pa[i] = p0v[i]; pa[8 + i] = p1v[i]; }
    }
    __builtin_amdgcn_sched_barrier(0);   // keep vf batch intact before chain

    // ---- O += P V : 8 WMMAs back-to-back on prefetched V fragments
#pragma unroll
    for (int j = 0; j < 8; ++j)
      acc[j] = wmma_f16(pa, vf[j], acc[j]);
  }

  // ---- epilogue: divide by row sums, store fp32 [B,T,HS]
#pragma unroll
  for (int r = 0; r < 8; ++r) {
    const float inv = 1.0f / lstate[r];
#pragma unroll
    for (int j = 0; j < 8; ++j)
      out[(rowQ + 8 * hh + r) * HS_ + j * 16 + m] = acc[j][r] * inv;
  }
}

// ---------------------------------------------------------------------------
extern "C" void kernel_launch(void* const* d_in, const int* in_sizes, int n_in,
                              void* d_out, int out_size, void* d_ws, size_t ws_size,
                              hipStream_t stream) {
  (void)in_sizes; (void)n_in; (void)out_size; (void)ws_size;
  const float* x  = (const float*)d_in[0];
  const float* Wk = (const float*)d_in[1];
  const float* Wq = (const float*)d_in[2];
  const float* Wv = (const float*)d_in[3];

  // Workspace: Q f16 [M,HS] | K f16 [M,HS] | Vt f16 [HS,M]  (12 MB total)
  _Float16* Qh = (_Float16*)d_ws;
  _Float16* Kh = Qh + (size_t)M_ * HS_;
  _Float16* Vt = Kh + (size_t)M_ * HS_;

  const float qscale = 0.08838834764831845f;   // HS^-0.5, folded into Q

  proj_gemm_kernel<<<M_ / 128, 256, 0, stream>>>(x, Wq, Qh, qscale, 0);
  proj_gemm_kernel<<<M_ / 128, 256, 0, stream>>>(x, Wk, Kh, 1.0f,   0);
  proj_gemm_kernel<<<M_ / 128, 256, 0, stream>>>(x, Wv, Vt, 1.0f,   1);
  attn_kernel<<<B_ * (T_ / 128), 256, 0, stream>>>(Qh, Kh, Vt, (float*)d_out);
}
